// ECGCLayer_16226386444552
// MI455X (gfx1250) — matrix operand order repaired and checked by reading, not compiled, not run on hardware
//
#include <hip/hip_runtime.h>

#define Dh 128
#define Eh 64
#define NN 256

typedef __attribute__((ext_vector_type(16))) _Float16 v16h;
typedef __attribute__((ext_vector_type(8)))  float    v8f;

__device__ __forceinline__ float fast_sigmoid_pre(float x, float nbl) {
    // sigmoid(x + bias) with nbl = -bias*log2(e):
    //   1 / (1 + exp2(-log2(e)*x + nbl))
    float e = __builtin_amdgcn_exp2f(__builtin_fmaf(x, -1.442695041f, nbl));
    return __builtin_amdgcn_rcpf(1.0f + e);
}

// ---------------------------------------------------------------------------
// Kernel 0: pack W2 (f32, [64][128] row-major) into B-operand-friendly f16
// layout Bt[d][k] (= W2^T), 128 rows x 64 halves.
// ---------------------------------------------------------------------------
__global__ void w2_pack_kernel(const float* __restrict__ W2,
                               _Float16* __restrict__ Bt) {
    int t = blockIdx.x * blockDim.x + threadIdx.x;      // 0..8191
    if (t < Dh * Eh) {
        int d = t >> 6;          // 0..127
        int k = t & 63;          // 0..63
        Bt[t] = (_Float16)W2[(size_t)k * Dh + d];
    }
}

// ---------------------------------------------------------------------------
// Fully fused ECGC layer: edge-gate MLP (WMMA) + masked mean/max aggregation
// + node-update MLP + residual + LayerNorm. One 256-thread block (8 waves)
// per (b,i) row; wave w owns gate columns d in [16w, 16w+16).
// ---------------------------------------------------------------------------
__global__ __launch_bounds__(256, 4)
void ecgc_fused_kernel(const float* __restrict__ h,
                       const float* __restrict__ ef,
                       const unsigned char* __restrict__ adj,
                       const float* __restrict__ W1,
                       const float* __restrict__ b1,
                       const _Float16* __restrict__ Bt,   // packed W2^T f16
                       const float* __restrict__ b2,
                       const float* __restrict__ U1, const float* __restrict__ b3,
                       const float* __restrict__ U2, const float* __restrict__ b4,
                       const float* __restrict__ gamma,
                       const float* __restrict__ beta,
                       float* __restrict__ out) {
    __shared__ float hs[Dh];
    __shared__ float base1p[4][Eh];
    __shared__ float base1[Eh];
    __shared__ float w1e[3][Eh];
    __shared__ float efs[NN * 3];
    __shared__ unsigned int maskw[8];
    __shared__ __attribute__((aligned(16))) _Float16 tA[NN][Eh];  // 32 KB
    // node-update stage
    __shared__ float uin[3 * Dh];
    __shared__ float part[2][Dh];
    __shared__ float r1[Dh];
    __shared__ float xs[Dh];
    __shared__ float red[Dh];
    __shared__ float red2[Dh];

    const int tid = threadIdx.x;
    const int bi  = blockIdx.x;                 // 0..B*N-1
    const size_t hoff = (size_t)bi * Dh;
    const float* __restrict__ hrow  = h  + hoff;
    const float* __restrict__ efrow = ef + (size_t)bi * NN * 3;
    const unsigned char* __restrict__ arow = adj + (size_t)bi * NN;

    // Warm L2/WGP$ for weight bytes used later (global_prefetch_b8).
    __builtin_prefetch(Bt + (size_t)(tid >> 5) * 16 * Eh, 0, 3);
    __builtin_prefetch(U1 + (size_t)tid * 192, 0, 2);

    // ---- stage inputs into LDS ----
    if (tid < Dh) hs[tid] = hrow[tid];
    if (tid < 3 * Eh) {                          // edge-feature rows of W1
        int r = tid / Eh, k = tid % Eh;
        w1e[r][k] = W1[(size_t)(Dh + r) * Eh + k];
    }
    for (int t = tid; t < NN * 3; t += 256) efs[t] = efrow[t];
    if (tid < 8) {
        unsigned int w = 0u;
        #pragma unroll
        for (int b = 0; b < 32; ++b)
            w |= (arow[tid * 32 + b] ? 1u : 0u) << b;
        maskw[tid] = w;
    }
    __syncthreads();

    // ---- base1[k] = h . W1[:128, k] + b1[k]  (h_j is constant over j!) ----
    {
        const int k = tid & 63;
        const int q = tid >> 6;                  // 0..3 -> d-chunk of 32
        float acc = 0.f;
        #pragma unroll 8
        for (int dd = 0; dd < 32; ++dd) {
            int d = 32 * q + dd;
            acc += hs[d] * W1[(size_t)d * Eh + k];
        }
        base1p[q][k] = acc;
    }
    __syncthreads();
    if (tid < Eh)
        base1[tid] = b1[tid] + base1p[0][tid] + base1p[1][tid] +
                     base1p[2][tid] + base1p[3][tid];
    __syncthreads();

    // ---- A matrix: t[j][k] = relu(base1[k] + ef[j].W1e[:,k]), f16 in LDS --
    {
        const int kg = tid & 15;                 // k-group of 4
        const int jb = tid >> 4;                 // 0..15
        const int k0 = 4 * kg;
        float w0a = w1e[0][k0], w0b = w1e[0][k0+1], w0c = w1e[0][k0+2], w0d = w1e[0][k0+3];
        float w1a = w1e[1][k0], w1b = w1e[1][k0+1], w1c = w1e[1][k0+2], w1d = w1e[1][k0+3];
        float w2a = w1e[2][k0], w2b = w1e[2][k0+1], w2c = w1e[2][k0+2], w2d = w1e[2][k0+3];
        float ba  = base1[k0], bb = base1[k0+1], bc = base1[k0+2], bd = base1[k0+3];
        #pragma unroll 4
        for (int it = 0; it < 16; ++it) {
            int j = jb + 16 * it;
            float e0 = efs[j * 3 + 0], e1 = efs[j * 3 + 1], e2 = efs[j * 3 + 2];
            float ta = fmaxf(ba + e0 * w0a + e1 * w1a + e2 * w2a, 0.f);
            float tb = fmaxf(bb + e0 * w0b + e1 * w1b + e2 * w2b, 0.f);
            float tc = fmaxf(bc + e0 * w0c + e1 * w1c + e2 * w2c, 0.f);
            float td = fmaxf(bd + e0 * w0d + e1 * w1d + e2 * w2d, 0.f);
            auto p0 = __builtin_amdgcn_cvt_pkrtz(ta, tb);   // v_cvt_pk_rtz_f16_f32
            auto p1 = __builtin_amdgcn_cvt_pkrtz(tc, td);
            uint2 u;
            __builtin_memcpy(&u.x, &p0, 4);
            __builtin_memcpy(&u.y, &p1, 4);
            *(uint2*)&tA[j][k0] = u;
        }
    }
    __syncthreads();

    // ---- WMMA GEMM (256x64)x(64x128) + fused sigmoid/mask/reduce ----
    const int lane = tid & 31;
    const int n    = lane & 15;       // column within tile / A row within tile
    const int hi   = lane >> 4;       // lane-half selects K (A,B) or M+8 (C)
    const int d0   = (tid >> 5) * 16; // wave's column base

    // B operands: lane holds Bt[d0+n][8*hi + 0..7] and [16 + 8*hi + 0..7] (table order)
    v16h Bk0, Bk1;
    {
        const _Float16* bp = Bt + (size_t)(d0 + n) * Eh + 8 * hi;
        ((uint4*)&Bk0)[0] = *(const uint4*)(bp + 0);
        ((uint4*)&Bk0)[1] = *(const uint4*)(bp + 16);
        ((uint4*)&Bk1)[0] = *(const uint4*)(bp + 32);
        ((uint4*)&Bk1)[1] = *(const uint4*)(bp + 48);
    }
    const float bias = b2[d0 + n];
    const float nbl  = -bias * 1.442695041f;     // fold bias into exp2 fma
    const float hd   = hs[d0 + n];

    const float INF = __builtin_inff();
    float sumg = 0.f, maxg = -INF, ming = INF;

    #pragma unroll 2
    for (int jt = 0; jt < 16; ++jt) {
        // A operands from LDS: row m = n, halves at 8*hi / +16 / +32 / +48
        v16h A0, A1;
        const _Float16* ta = &tA[jt * 16 + n][0];
        ((uint4*)&A0)[0] = *(const uint4*)(ta +  0 + 8 * hi);
        ((uint4*)&A0)[1] = *(const uint4*)(ta + 16 + 8 * hi);
        ((uint4*)&A1)[0] = *(const uint4*)(ta + 32 + 8 * hi);
        ((uint4*)&A1)[1] = *(const uint4*)(ta + 48 + 8 * hi);

        v8f c = {};
        c = __builtin_amdgcn_wmma_f32_16x16x32_f16(false, A0, false, Bk0,
                                                   (short)0, c, false, false);
        c = __builtin_amdgcn_wmma_f32_16x16x32_f16(false, A1, false, Bk1,
                                                   (short)0, c, false, false);

        // C layout: VGPR v, lane-half hi -> row j = 16*jt + v + 8*hi
        unsigned int mw = maskw[jt >> 1] >> ((jt & 1) * 16);
        #pragma unroll
        for (int v = 0; v < 8; ++v) {
            float g = fast_sigmoid_pre(c[v], nbl);
            bool m = (mw >> (v + 8 * hi)) & 1u;
            sumg += m ? g : 0.0f;
            maxg  = fmaxf(maxg, m ? g : -INF);
            ming  = fminf(ming, m ? g :  INF);
        }
    }
    // combine the two lane halves (same column d, disjoint row sets)
    sumg +=        __shfl_xor(sumg, 16, 32);
    maxg  = fmaxf(maxg, __shfl_xor(maxg, 16, 32));
    ming  = fminf(ming, __shfl_xor(ming, 16, 32));

    int cnt = 0;
    #pragma unroll
    for (int w = 0; w < 8; ++w) cnt += __popc(maskw[w]);

    if (hi == 0) {
        float am, ax;
        if (cnt == 0) { am = 0.f; ax = 0.f; }
        else {
            am = sumg * hd * __builtin_amdgcn_rcpf((float)cnt);  // cnt >= 1
            ax = (hd >= 0.f ? maxg : ming) * hd;                 // fixed sign
        }
        uin[Dh + d0 + n]     = am;
        uin[2 * Dh + d0 + n] = ax;
    }
    if (tid < Dh) uin[tid] = hs[tid];
    __syncthreads();

    // ---- node update MLP: relu(uin @ U1 + b3) @ U2 + b4, fused in-block ----
    const int d = tid & 127;
    const int q = tid >> 7;                      // thread-half: e-range split
    {
        float acc = 0.f;
        #pragma unroll 8
        for (int ee = 0; ee < 192; ++ee) {
            int e = 192 * q + ee;
            acc += uin[e] * U1[(size_t)e * Dh + d];
        }
        part[q][d] = acc;
    }
    __syncthreads();
    if (tid < Dh) r1[tid] = fmaxf(b3[tid] + part[0][tid] + part[1][tid], 0.f);
    __syncthreads();
    {
        float acc = 0.f;
        #pragma unroll 8
        for (int ee = 0; ee < 64; ++ee) {
            int e = 64 * q + ee;
            acc += r1[e] * U2[(size_t)e * Dh + d];
        }
        part[q][d] = acc;
    }
    __syncthreads();

    // ---- residual + LayerNorm ----
    if (tid < Dh) {
        float x = hs[tid] + b4[tid] + part[0][tid] + part[1][tid];
        xs[tid] = x;
        red[tid] = x;
        red2[tid] = x * x;
    }
    __syncthreads();
    for (int s = Dh / 2; s > 0; s >>= 1) {
        if (tid < s) { red[tid] += red[tid + s]; red2[tid] += red2[tid + s]; }
        __syncthreads();
    }
    if (tid < Dh) {
        float x   = xs[tid];
        float mu  = red[0]  * (1.0f / Dh);
        float var = red2[0] * (1.0f / Dh) - mu * mu;
        out[hoff + tid] = (x - mu) * rsqrtf(var + 1e-5f) * gamma[tid] + beta[tid];
    }
}

// ---------------------------------------------------------------------------
extern "C" void kernel_launch(void* const* d_in, const int* in_sizes, int n_in,
                              void* d_out, int out_size, void* d_ws, size_t ws_size,
                              hipStream_t stream) {
    const float*         h     = (const float*)d_in[0];
    const float*         ef    = (const float*)d_in[1];
    const unsigned char* adj   = (const unsigned char*)d_in[2];
    const float*         W1    = (const float*)d_in[3];
    const float*         b1    = (const float*)d_in[4];
    const float*         W2    = (const float*)d_in[5];
    const float*         b2    = (const float*)d_in[6];
    const float*         U1    = (const float*)d_in[7];
    const float*         b3    = (const float*)d_in[8];
    const float*         U2    = (const float*)d_in[9];
    const float*         b4    = (const float*)d_in[10];
    const float*         gamma = (const float*)d_in[11];
    const float*         beta  = (const float*)d_in[12];
    float*               out   = (float*)d_out;

    const int BN = in_sizes[0] / Dh;            // B*N = 2048

    _Float16* Bt = (_Float16*)d_ws;             // 16 KB packed W2^T (f16)

    w2_pack_kernel<<<(Dh * Eh + 255) / 256, 256, 0, stream>>>(W2, Bt);
    ecgc_fused_kernel<<<BN, 256, 0, stream>>>(h, ef, adj, W1, b1, Bt, b2,
                                              U1, b3, U2, b4, gamma, beta, out);
}
